// TSPNet_18957985644718
// MI455X (gfx1250) — matrix operand order repaired and checked by reading, not compiled, not run on hardware
//
#include <hip/hip_runtime.h>
#include <hip/hip_bf16.h>

typedef __attribute__((ext_vector_type(16))) _Float16 v16h;
typedef __attribute__((ext_vector_type(8)))  _Float16 v8h;
typedef __attribute__((ext_vector_type(8)))  float    v8f;

#define NBATCH 512
#define SEQL   128
#define HID    512
#define G4     2048
#define TSTEPS 129   // L+1 decode steps
#define NTHR   512   // 16 waves per workgroup (wave32)

// ---------------------------------------------------------------------------
// WMMA helpers (CDNA5 gfx1250, wave32)
// ---------------------------------------------------------------------------
__device__ __forceinline__ v8f wmma_f16(v16h a, v16h b, v8f c) {
  // D(16x16 f32) = A(16x32 f16) * B(32x16 f16) + C
  return __builtin_amdgcn_wmma_f32_16x16x32_f16(
      /*neg_a=*/false, a, /*neg_b=*/false, b,
      /*c_mod=*/(short)0, c, /*reuse_a=*/false, /*reuse_b=*/false);
}

// A fragment 16x32 f16 from a row-major tile with row stride 512 elements.
// base points at (row 0, k0). Per ISA: lanes 0-15 hold row=lane, K = 0..7 and
// 16..23; lanes 16-31 hold row=lane-16, K = 8..15 and 24..31.
__device__ __forceinline__ v16h load_a512(const _Float16* base, int lane) {
  const _Float16* p = base + (lane & 15) * HID + ((lane >> 4) << 3);
  v8h lo = *(const v8h*)(p);
  v8h hi = *(const v8h*)(p + 16);
  v16h r;
#pragma unroll
  for (int i = 0; i < 8; ++i) { r[i] = lo[i]; r[i + 8] = hi[i]; }
  return r;
}

// B fragment 32x16 f16 where B(k, c) = W[colBase + c][k0 + k], W row-major
// [nout][ldw]. Per ISA: lanes 0-15 hold col=lane, K=0..7 and 16..23;
// lanes 16-31 hold col=lane-16, K=8..15 and 24..31.
__device__ __forceinline__ v16h load_b(const _Float16* w, int ldw,
                                       int colBase, int k0, int lane) {
  const _Float16* p = w + (size_t)(colBase + (lane & 15)) * ldw + k0 + ((lane >> 4) << 3);
  v8h lo = *(const v8h*)(p);
  v8h hi = *(const v8h*)(p + 16);
  v16h r;
#pragma unroll
  for (int i = 0; i < 8; ++i) { r[i] = lo[i]; r[i + 8] = hi[i]; }
  return r;
}

// ---------------------------------------------------------------------------
// Math helpers
// ---------------------------------------------------------------------------
__device__ __forceinline__ float sigm(float x) {
  x = fminf(fmaxf(x, -30.f), 30.f);
  return 1.0f / (1.0f + __expf(-x));
}
__device__ __forceinline__ float tanh_fast(float x) {
  x = fminf(fmaxf(x, -15.f), 15.f);
  float e = __expf(2.f * x);
  return (e - 1.f) / (e + 1.f);
}
__device__ __forceinline__ float wave_max(float v) {
#pragma unroll
  for (int m = 16; m > 0; m >>= 1) v = fmaxf(v, __shfl_xor(v, m, 32));
  return v;
}
__device__ __forceinline__ float wave_sum(float v) {
#pragma unroll
  for (int m = 16; m > 0; m >>= 1) v += __shfl_xor(v, m, 32);
  return v;
}

// ---------------------------------------------------------------------------
// LSTM gate GEMM: gates(16x2048) = xs(16x512) @ Wih^T + hs(16x512) @ Whh^T.
// 16 waves; wave owns hidden units [wave*32, wave*32+32): 2 column tiles per
// gate (i,f,g,o tiles share the lane<->(row,col) mapping, so the cell update
// runs elementwise on accumulators). 64 accumulator VGPRs per wave keeps the
// wave below 256 VGPRs -> no s_set_vgpr_msb churn, and 4 waves/SIMD hide the
// L2 weight-stream latency.
// ---------------------------------------------------------------------------
__device__ __forceinline__ void lstm_gates(const _Float16* xs, const _Float16* hs,
                                           const _Float16* Wih, const _Float16* Whh,
                                           int wave, int lane, v8f acc[4][2]) {
#pragma unroll
  for (int g = 0; g < 4; ++g)
#pragma unroll
    for (int q = 0; q < 2; ++q)
#pragma unroll
      for (int e = 0; e < 8; ++e) acc[g][q][e] = 0.f;

  for (int k0 = 0; k0 < HID; k0 += 32) {
    v16h a = load_a512(xs + k0, lane);
#pragma unroll
    for (int g = 0; g < 4; ++g)
#pragma unroll
      for (int q = 0; q < 2; ++q) {
        v16h b = load_b(Wih, HID, g * HID + wave * 32 + q * 16, k0, lane);
        acc[g][q] = wmma_f16(a, b, acc[g][q]);
      }
  }
  for (int k0 = 0; k0 < HID; k0 += 32) {
    v16h a = load_a512(hs + k0, lane);
#pragma unroll
    for (int g = 0; g < 4; ++g)
#pragma unroll
      for (int q = 0; q < 2; ++q) {
        v16h b = load_b(Whh, HID, g * HID + wave * 32 + q * 16, k0, lane);
        acc[g][q] = wmma_f16(a, b, acc[g][q]);
      }
  }
}

// ---------------------------------------------------------------------------
// Weight f32 -> f16 cast (layout preserved: rows already match B-fragment
// contiguity requirement).
// ---------------------------------------------------------------------------
__global__ void cvt_f32_f16(const float* __restrict__ s, _Float16* __restrict__ d, int n) {
  int i = blockIdx.x * blockDim.x + threadIdx.x;
  if (i < n) d[i] = (_Float16)s[i];
}

// ---------------------------------------------------------------------------
// Encoder: fc + 3-layer LSTM over all 128 steps + K projection.
// One workgroup = 16 batch rows, fully private recurrence (no grid sync).
// ---------------------------------------------------------------------------
struct EncArgs {
  const float* X; const float* fcw; const float* fcb;
  const _Float16* Wih[3]; const _Float16* Whh[3];
  const float* bih[3]; const float* bhh[3];
  const _Float16* Wk; const float* bk;
  _Float16* bufA; _Float16* bufB; _Float16* kbuf;
};

__global__ __launch_bounds__(NTHR) void enc_kernel(EncArgs A) {
  __shared__ _Float16 xs[16 * HID];     // 16 KB  step input (f16, A-source)
  __shared__ _Float16 hs[16 * HID];     // 16 KB  recurrent h (f16, A-source)
  __shared__ float    biasL[G4];        // 8 KB   bih+bhh of current layer
  __shared__ float    fcwL[HID * 2];    // 4 KB
  __shared__ float    fcbL[HID];        // 2 KB

  const int tid  = threadIdx.x;
  const int lane = tid & 31;
  const int wave = tid >> 5;
  const int n0   = blockIdx.x * 16;

  for (int i = tid; i < HID * 2; i += NTHR) fcwL[i] = A.fcw[i];
  for (int i = tid; i < HID; i += NTHR) fcbL[i] = A.fcb[i];
  __syncthreads();

  for (int l = 0; l < 3; ++l) {
    const _Float16* inBuf = (l == 1) ? A.bufA : A.bufB;  // l1 reads A, l2 reads B
    _Float16* outBuf = (l == 1) ? A.bufB : A.bufA;       // l0->A, l1->B, l2->A

    for (int i = tid; i < G4; i += NTHR) biasL[i] = A.bih[l][i] + A.bhh[l][i];
    for (int i = tid; i < 16 * HID; i += NTHR) hs[i] = (_Float16)0.f;
    __syncthreads();

    v8f c[2];
#pragma unroll
    for (int q = 0; q < 2; ++q)
#pragma unroll
      for (int e = 0; e < 8; ++e) c[q][e] = 0.f;

    for (int t = 0; t < SEQL; ++t) {
      // stage step input
      if (l == 0) {
        for (int i = tid; i < 16 * HID; i += NTHR) {
          int row = i >> 9, h = i & 511;
          const float* xp = A.X + ((size_t)(n0 + row) * SEQL + t) * 2;
          xs[i] = (_Float16)(xp[0] * fcwL[h * 2 + 0] + xp[1] * fcwL[h * 2 + 1] + fcbL[h]);
        }
      } else {
        for (int i = tid; i < 16 * HID; i += NTHR) {
          int row = i >> 9, h = i & 511;
          xs[i] = inBuf[((size_t)(n0 + row) * SEQL + t) * HID + h];
        }
      }
      __syncthreads();

      v8f acc[4][2];
      lstm_gates(xs, hs, A.Wih[l], A.Whh[l], wave, lane, acc);
      __syncthreads();

      // cell update elementwise on accumulators; c lives in VGPRs
#pragma unroll
      for (int q = 0; q < 2; ++q) {
        int col = wave * 32 + q * 16 + (lane & 15);
        float bI = biasL[col], bF = biasL[HID + col];
        float bG = biasL[2 * HID + col], bO = biasL[3 * HID + col];
#pragma unroll
        for (int r = 0; r < 8; ++r) {
          float iv = sigm(acc[0][q][r] + bI);
          float fv = sigm(acc[1][q][r] + bF);
          float gv = tanh_fast(acc[2][q][r] + bG);
          float ov = sigm(acc[3][q][r] + bO);
          float cn = fv * c[q][r] + iv * gv;
          c[q][r] = cn;
          float hv = ov * tanh_fast(cn);
          int row = r + ((lane >> 4) << 3);
          hs[row * HID + col] = (_Float16)hv;
          outBuf[((size_t)(n0 + row) * SEQL + t) * HID + col] = (_Float16)hv;
        }
      }
      __syncthreads();
    }
    __syncthreads();
  }

  // K projection: k = enc @ Wk^T + bk  (enc final is bufA, f16)
  __threadfence_block();
  __syncthreads();
  for (int mi = 0; mi < 8; ++mi) {
    int mt = wave * 8 + mi;          // 128 M-tiles over 16 waves
    int nl = mt >> 3, tt = mt & 7;   // 16 seq rows per A tile, fixed batch row
    const _Float16* abase = A.bufA + ((size_t)(n0 + nl) * SEQL + tt * 16) * HID;
    for (int ct = 0; ct < 32; ++ct) {
      v8f acc;
#pragma unroll
      for (int e = 0; e < 8; ++e) acc[e] = 0.f;
      for (int k0 = 0; k0 < HID; k0 += 32) {
        v16h a = load_a512(abase + k0, lane);
        v16h b = load_b(A.Wk, HID, ct * 16, k0, lane);
        acc = wmma_f16(a, b, acc);
      }
      int col = ct * 16 + (lane & 15);
      float bv = A.bk[col];
#pragma unroll
      for (int r = 0; r < 8; ++r) {
        int trow = tt * 16 + r + ((lane >> 4) << 3);
        A.kbuf[((size_t)(n0 + nl) * SEQL + trow) * HID + col] = (_Float16)(acc[r] + bv);
      }
    }
  }
}

// ---------------------------------------------------------------------------
// Decoder: all 129 steps fused in one kernel. Per-WG state (h,c,mask,first)
// is private to its 16 batch rows.
// ---------------------------------------------------------------------------
struct DecArgs {
  const _Float16* Wih[3]; const _Float16* Whh[3];
  const float* bih[3]; const float* bhh[3];
  const _Float16* Wq; const float* bq;
  const _Float16* enc;   // (N,L,H) f16
  const _Float16* kbuf;  // (N,L,H) f16
  float* outAttn;        // (N,129,128)
  float* outIdx;         // (N,129)
};

__global__ __launch_bounds__(NTHR) void dec_kernel(DecArgs A) {
  extern __shared__ char smem[];
  _Float16* xstar = (_Float16*)smem;                 // 16x512 f16   16 KB
  _Float16* hbuf  = xstar + 16 * HID;                // 3x16x512 f16 48 KB
  _Float16* qls   = hbuf + 3 * 16 * HID;             // 16x512 f16   16 KB
  float* cbuf     = (float*)(qls + 16 * HID);        // 3x16x512 f32 96 KB
  float* biasD    = cbuf + 3 * 16 * HID;             // 3x2048 f32   24 KB
  float* scoresL  = biasD + 3 * G4;                  // 16x256 f32   16 KB
  float* probsL   = scoresL + 16 * 256;              // 16x256 f32   16 KB
  unsigned char* maskL = (unsigned char*)(probsL + 16 * 256); // 2 KB
  int* firstL = (int*)(maskL + 16 * SEQL);
  int* idxL   = firstL + 16;

  const int tid  = threadIdx.x;
  const int lane = tid & 31;
  const int wave = tid >> 5;
  const int n0   = blockIdx.x * 16;

  for (int i = tid; i < 16 * HID; i += NTHR) xstar[i] = (_Float16)0.f;
  for (int i = tid; i < 3 * 16 * HID; i += NTHR) { hbuf[i] = (_Float16)0.f; cbuf[i] = 0.f; }
  for (int i = tid; i < 3 * G4; i += NTHR) {
    int l = i / G4, j = i - l * G4;
    biasD[i] = A.bih[l][j] + A.bhh[l][j];
  }
  for (int i = tid; i < 16 * SEQL; i += NTHR) maskL[i] = 0;
  if (tid < 16) { firstL[tid] = 0; idxL[tid] = 0; }
  __syncthreads();

  for (int t = 0; t <= SEQL; ++t) {
    // ---- 3 decoder LSTM cells ----
    for (int l = 0; l < 3; ++l) {
      const _Float16* xsP = (l == 0) ? xstar : (hbuf + (l - 1) * 16 * HID);
      _Float16* hsP = hbuf + l * 16 * HID;
      float* cP = cbuf + l * 16 * HID;
      v8f acc[4][2];
      lstm_gates(xsP, hsP, A.Wih[l], A.Whh[l], wave, lane, acc);
      __syncthreads();
#pragma unroll
      for (int q = 0; q < 2; ++q) {
        int col = wave * 32 + q * 16 + (lane & 15);
        const float* bl = biasD + l * G4;
        float bI = bl[col], bF = bl[HID + col];
        float bG = bl[2 * HID + col], bO = bl[3 * HID + col];
#pragma unroll
        for (int r = 0; r < 8; ++r) {
          int row = r + ((lane >> 4) << 3);
          float cprev = cP[row * HID + col];
          float iv = sigm(acc[0][q][r] + bI);
          float fv = sigm(acc[1][q][r] + bF);
          float gv = tanh_fast(acc[2][q][r] + bG);
          float ov = sigm(acc[3][q][r] + bO);
          float cn = fv * cprev + iv * gv;
          cP[row * HID + col] = cn;
          hsP[row * HID + col] = (_Float16)(ov * tanh_fast(cn));
        }
      }
      __syncthreads();
    }

    // ---- q = (y @ Wq^T + bq) / sqrt(hd) ----
    {
      const _Float16* y = hbuf + 2 * 16 * HID;
#pragma unroll
      for (int cti = 0; cti < 2; ++cti) {
        int ct = wave * 2 + cti;    // 32 column tiles over 16 waves
        v8f acc;
#pragma unroll
        for (int e = 0; e < 8; ++e) acc[e] = 0.f;
        for (int k0 = 0; k0 < HID; k0 += 32) {
          v16h a = load_a512(y + k0, lane);
          v16h b = load_b(A.Wq, HID, ct * 16, k0, lane);
          acc = wmma_f16(a, b, acc);
        }
        int col = ct * 16 + (lane & 15);
        float bv = A.bq[col];
#pragma unroll
        for (int r = 0; r < 8; ++r) {
          int row = r + ((lane >> 4) << 3);
          qls[row * HID + col] = (_Float16)((acc[r] + bv) * 0.0625f); // 1/sqrt(256)
        }
      }
    }
    __syncthreads();

    // ---- attention scores: per-batch-row GEMV (M=1 -> VALU, ~1% of FLOPs) ----
    for (int it = 0; it < 8; ++it) {
      int task = it * NTHR + tid;          // 16 rows x 2 heads x 128 keys
      int row  = task >> 8;
      int head = (task >> 7) & 1;
      int key  = task & 127;
      const v8h* qp = (const v8h*)(qls + row * HID + head * 256);
      const v8h* kp = (const v8h*)(A.kbuf + ((size_t)(n0 + row) * SEQL + key) * HID + head * 256);
      float s = 0.f;
#pragma unroll 4
      for (int cidx = 0; cidx < 32; ++cidx) {
        v8h qv = qp[cidx]; v8h kv = kp[cidx];
#pragma unroll
        for (int e = 0; e < 8; ++e) s += (float)qv[e] * (float)kv[e];
      }
      scoresL[row * 256 + head * 128 + key] = s;
    }
    __syncthreads();

    // ---- masked per-head softmax (32 (row,head) tasks over 16 waves) ----
#pragma unroll
    for (int i = 0; i < 2; ++i) {
      int task = wave * 2 + i;
      int row = task >> 1, head = task & 1;
      float s[4];
#pragma unroll
      for (int j = 0; j < 4; ++j) {
        int key = lane + j * 32;
        bool m = maskL[row * SEQL + key] != 0;
        if (t == SEQL && key == firstL[row]) m = false;  // tour closure
        s[j] = m ? -1e9f : scoresL[row * 256 + head * 128 + key];
      }
      float mx = wave_max(fmaxf(fmaxf(s[0], s[1]), fmaxf(s[2], s[3])));
      float e[4]; float ps = 0.f;
#pragma unroll
      for (int j = 0; j < 4; ++j) { e[j] = __expf(s[j] - mx); ps += e[j]; }
      ps = wave_sum(ps);
      float inv = 1.0f / ps;
#pragma unroll
      for (int j = 0; j < 4; ++j)
        probsL[row * 256 + head * 128 + lane + j * 32] = e[j] * inv;
    }
    __syncthreads();

    // ---- head average, clip+renorm, write attn, argmax, state update ----
    {
      int row = wave;  // 16 rows over 16 waves
      float a[4]; float psum = 0.f;
#pragma unroll
      for (int j = 0; j < 4; ++j) {
        int key = lane + j * 32;
        float v = 0.5f * (probsL[row * 256 + key] + probsL[row * 256 + 128 + key]);
        v = fmaxf(v, 1e-9f);
        a[j] = v; psum += v;
      }
      psum = wave_sum(psum);
      float inv = 1.0f / psum;
      float bv = -1.f; int bi = 0;
      float* orow = A.outAttn + ((size_t)(n0 + row) * TSTEPS + t) * SEQL;
#pragma unroll
      for (int j = 0; j < 4; ++j) {
        int key = lane + j * 32;
        float v = a[j] * inv;
        orow[key] = v;
        if (v > bv) { bv = v; bi = key; }
      }
#pragma unroll
      for (int m = 16; m > 0; m >>= 1) {
        float ov = __shfl_xor(bv, m, 32);
        int   oi = __shfl_xor(bi, m, 32);
        if (ov > bv || (ov == bv && oi < bi)) { bv = ov; bi = oi; }
      }
      if (lane == 0) {
        idxL[row] = bi;
        maskL[row * SEQL + bi] = 1;
        if (t == 0) firstL[row] = bi;
        A.outIdx[(size_t)(n0 + row) * TSTEPS + t] = (float)bi;
      }
    }
    __syncthreads();

    // ---- gather x_star = enc[n, idx] ----
    for (int i = tid; i < 16 * (HID / 8); i += NTHR) {
      int row = i >> 6, ch = i & 63;
      const v8h* src = (const v8h*)(A.enc + ((size_t)(n0 + row) * SEQL + idxL[row]) * HID);
      ((v8h*)(xstar + row * HID))[ch] = src[ch];
    }
    __syncthreads();
  }
}

// ---------------------------------------------------------------------------
// Host launch
// ---------------------------------------------------------------------------
extern "C" void kernel_launch(void* const* d_in, const int* in_sizes, int n_in,
                              void* d_out, int out_size, void* d_ws, size_t ws_size,
                              hipStream_t stream) {
  (void)in_sizes; (void)n_in; (void)out_size; (void)ws_size;

  const float* X   = (const float*)d_in[0];
  const float* fcw = (const float*)d_in[1];
  const float* fcb = (const float*)d_in[2];
  const float* encW[3][4]; const float* decW[3][4];
  for (int l = 0; l < 3; ++l)
    for (int k = 0; k < 4; ++k) {
      encW[l][k] = (const float*)d_in[3 + 4 * l + k];
      decW[l][k] = (const float*)d_in[15 + 4 * l + k];
    }
  const float* Wq = (const float*)d_in[27];
  const float* bq = (const float*)d_in[28];
  const float* Wk = (const float*)d_in[29];
  const float* bk = (const float*)d_in[30];

  char* ws = (char*)d_ws;
  size_t off = 0;
  auto alloc = [&](size_t bytes) {
    char* p = ws + off;
    off += (bytes + 255) & ~(size_t)255;
    return p;
  };
  const size_t WSZ = (size_t)G4 * HID * sizeof(_Float16);
  _Float16 *wEncIh[3], *wEncHh[3], *wDecIh[3], *wDecHh[3];
  for (int l = 0; l < 3; ++l) { wEncIh[l] = (_Float16*)alloc(WSZ); wEncHh[l] = (_Float16*)alloc(WSZ); }
  for (int l = 0; l < 3; ++l) { wDecIh[l] = (_Float16*)alloc(WSZ); wDecHh[l] = (_Float16*)alloc(WSZ); }
  _Float16* wQ = (_Float16*)alloc((size_t)HID * HID * sizeof(_Float16));
  _Float16* wK = (_Float16*)alloc((size_t)HID * HID * sizeof(_Float16));
  const size_t ABSZ = (size_t)NBATCH * SEQL * HID * sizeof(_Float16);
  _Float16* bufA = (_Float16*)alloc(ABSZ);
  _Float16* bufB = (_Float16*)alloc(ABSZ);
  _Float16* kbuf = (_Float16*)alloc(ABSZ);

  // f32 -> f16 weight casts (layout preserved == WMMA B-fragment friendly)
  const int nW = G4 * HID;
  dim3 cb(256), cgW((nW + 255) / 256);
  for (int l = 0; l < 3; ++l) {
    cvt_f32_f16<<<cgW, cb, 0, stream>>>(encW[l][0], wEncIh[l], nW);
    cvt_f32_f16<<<cgW, cb, 0, stream>>>(encW[l][1], wEncHh[l], nW);
    cvt_f32_f16<<<cgW, cb, 0, stream>>>(decW[l][0], wDecIh[l], nW);
    cvt_f32_f16<<<cgW, cb, 0, stream>>>(decW[l][1], wDecHh[l], nW);
  }
  const int nQ = HID * HID;
  dim3 cgQ((nQ + 255) / 256);
  cvt_f32_f16<<<cgQ, cb, 0, stream>>>(Wq, wQ, nQ);
  cvt_f32_f16<<<cgQ, cb, 0, stream>>>(Wk, wK, nQ);

  EncArgs ea;
  ea.X = X; ea.fcw = fcw; ea.fcb = fcb;
  for (int l = 0; l < 3; ++l) {
    ea.Wih[l] = wEncIh[l]; ea.Whh[l] = wEncHh[l];
    ea.bih[l] = encW[l][2]; ea.bhh[l] = encW[l][3];
  }
  ea.Wk = wK; ea.bk = bk; ea.bufA = bufA; ea.bufB = bufB; ea.kbuf = kbuf;
  enc_kernel<<<dim3(NBATCH / 16), dim3(NTHR), 0, stream>>>(ea);

  DecArgs da;
  for (int l = 0; l < 3; ++l) {
    da.Wih[l] = wDecIh[l]; da.Whh[l] = wDecHh[l];
    da.bih[l] = decW[l][2]; da.bhh[l] = decW[l][3];
  }
  da.Wq = wQ; da.bq = bq; da.enc = bufA; da.kbuf = kbuf;
  da.outAttn = (float*)d_out;
  da.outIdx  = (float*)d_out + (size_t)NBATCH * TSTEPS * SEQL;

  const size_t decLds =
      (size_t)16 * HID * 2          // xstar
    + (size_t)3 * 16 * HID * 2      // hbuf
    + (size_t)16 * HID * 2          // qls
    + (size_t)3 * 16 * HID * 4      // cbuf
    + (size_t)3 * G4 * 4            // biasD
    + (size_t)16 * 256 * 4          // scoresL
    + (size_t)16 * 256 * 4          // probsL
    + (size_t)16 * SEQL             // maskL
    + 256;                          // firstL/idxL + pad
  dec_kernel<<<dim3(NBATCH / 16), dim3(NTHR), decLds, stream>>>(da);
}
